// Lambda_4423816315159
// MI455X (gfx1250) — compile-verified, matrix operand order checked by reading, not run on hardware
//
#include <hip/hip_runtime.h>
#include <hip/hip_bf16.h>

// ---------------------------------------------------------------------------
// Problem constants (from reference)
// ---------------------------------------------------------------------------
#define BB 8
#define NN 4096
#define DD 768
#define HH 12
#define KK 16
#define UU 4
#define VV 64
#define LL 23
#define PADW 11
#define M_TOT (BB * NN)          // 32768 flattened (b,n) positions
#define C_TOT 512                // 192 q + 64 k + 256 v channels
#define CQ0 0
#define CK0 192
#define CV0 256

typedef __attribute__((ext_vector_type(16))) _Float16     v16h;
typedef __attribute__((ext_vector_type(8)))  float        v8f;
typedef __attribute__((ext_vector_type(8)))  unsigned int v8u;

union Frag16 { v8u u; v16h h; };

// K index inside a 16-bit 16x32 WMMA A/B fragment (per CDNA5 ISA layout)
__device__ __forceinline__ int wmma_k_of(int lane, int i) {
    return 16 * (i >> 2) + 8 * (lane >> 4) + 2 * (i & 3);
}

// ---------------------------------------------------------------------------
// K1: convert x (f32) -> f16, row-major [m][d]
// ---------------------------------------------------------------------------
__global__ void cvt_x_kernel(const float* __restrict__ x, _Float16* __restrict__ xh) {
    size_t i = (size_t)blockIdx.x * 256 + threadIdx.x;   // grid sized exactly
    xh[i] = (_Float16)x[i];
}

// ---------------------------------------------------------------------------
// K2: pack Wq|Wk|Wv -> f16, row-major [c][d], c in [0,512)
// ---------------------------------------------------------------------------
__global__ void cvt_w_kernel(const float* __restrict__ Wq, const float* __restrict__ Wk,
                             const float* __restrict__ Wv, _Float16* __restrict__ wh) {
    int i = blockIdx.x * 256 + threadIdx.x;              // 512*768 total
    int c = i / DD, d = i % DD;
    float v;
    if (c < CK0)      v = Wq[c * DD + d];
    else if (c < CV0) v = Wk[(c - CK0) * DD + d];
    else              v = Wv[(c - CV0) * DD + d];
    wh[i] = (_Float16)v;
}

// ---------------------------------------------------------------------------
// K3: WMMA projection GEMM: qkv[c][m] = sum_d W[c][d] * X[m][d]
//     block = 128 threads = 4 waves; each wave: 16 c-rows x 64 m-cols
// ---------------------------------------------------------------------------
__global__ void gemm_qkv_kernel(const _Float16* __restrict__ W,
                                const _Float16* __restrict__ X,
                                float* __restrict__ qkv) {
    const int lane = threadIdx.x & 31;
    const int wave = threadIdx.x >> 5;
    const int c0 = blockIdx.x * 64 + wave * 16;   // channel tile
    const int m0 = blockIdx.y * 64;               // position tile (4 x 16)

    v8f acc0 = {}, acc1 = {}, acc2 = {}, acc3 = {};
    const int arow = c0 + (lane & 15);
    const int mbase = m0 + (lane & 15);

    for (int d0 = 0; d0 < DD; d0 += 32) {
        Frag16 a;
#pragma unroll
        for (int i = 0; i < 8; ++i) {
            int k = wmma_k_of(lane, i);
            a.u[i] = *(const unsigned int*)(W + (size_t)arow * DD + d0 + k);
        }
        Frag16 b0, b1, b2, b3;
#pragma unroll
        for (int i = 0; i < 8; ++i) {
            int k = wmma_k_of(lane, i);
            b0.u[i] = *(const unsigned int*)(X + (size_t)(mbase +  0) * DD + d0 + k);
            b1.u[i] = *(const unsigned int*)(X + (size_t)(mbase + 16) * DD + d0 + k);
            b2.u[i] = *(const unsigned int*)(X + (size_t)(mbase + 32) * DD + d0 + k);
            b3.u[i] = *(const unsigned int*)(X + (size_t)(mbase + 48) * DD + d0 + k);
        }
        acc0 = __builtin_amdgcn_wmma_f32_16x16x32_f16(false, a.h, false, b0.h, (short)0, acc0, false, false);
        acc1 = __builtin_amdgcn_wmma_f32_16x16x32_f16(false, a.h, false, b1.h, (short)0, acc1, false, false);
        acc2 = __builtin_amdgcn_wmma_f32_16x16x32_f16(false, a.h, false, b2.h, (short)0, acc2, false, false);
        acc3 = __builtin_amdgcn_wmma_f32_16x16x32_f16(false, a.h, false, b3.h, (short)0, acc3, false, false);
    }

    // C/D layout: lane l, vgpr j -> row c0 + j + 8*(l>>4), col m-tile + (l&15)
    const int crow = c0 + 8 * (lane >> 4);
    const int mcol = m0 + (lane & 15);
#pragma unroll
    for (int j = 0; j < 8; ++j) {
        qkv[(size_t)(crow + j) * M_TOT + mcol +  0] = acc0[j];
        qkv[(size_t)(crow + j) * M_TOT + mcol + 16] = acc1[j];
        qkv[(size_t)(crow + j) * M_TOT + mcol + 32] = acc2[j];
        qkv[(size_t)(crow + j) * M_TOT + mcol + 48] = acc3[j];
    }
}

// ---------------------------------------------------------------------------
// K4: BatchNorm stats for q (192 ch) and v (256 ch) -> folded scale/shift
//     ch in [0,192): q channel; ch in [192,448): v channel (cv = ch-192)
// ---------------------------------------------------------------------------
__global__ void bn_stats_kernel(const float* __restrict__ qkv,
                                const float* __restrict__ gq, const float* __restrict__ bq,
                                const float* __restrict__ gv, const float* __restrict__ bv,
                                float* __restrict__ scale, float* __restrict__ shift) {
    int ch = blockIdx.x;                                  // 0..447
    int c = (ch < 192) ? ch : (CV0 + (ch - 192));
    const float* row = qkv + (size_t)c * M_TOT;
    float s = 0.f, s2 = 0.f;
    for (int i = threadIdx.x; i < M_TOT; i += 256) {
        float x = row[i];
        s += x; s2 += x * x;
    }
    __shared__ float rs[256], rs2[256];
    rs[threadIdx.x] = s; rs2[threadIdx.x] = s2;
    __syncthreads();
    for (int o = 128; o > 0; o >>= 1) {
        if (threadIdx.x < o) { rs[threadIdx.x] += rs[threadIdx.x + o]; rs2[threadIdx.x] += rs2[threadIdx.x + o]; }
        __syncthreads();
    }
    if (threadIdx.x == 0) {
        float mean = rs[0] * (1.f / M_TOT);
        float var  = rs2[0] * (1.f / M_TOT) - mean * mean;
        float rstd = rsqrtf(var + 1e-5f);
        float g, b;
        if (ch < 192) { g = gq[ch]; b = bq[ch]; }
        else          { g = gv[ch - 192]; b = bv[ch - 192]; }
        scale[ch] = g * rstd;
        shift[ch] = b - mean * g * rstd;
    }
}

// ---------------------------------------------------------------------------
// K5: softmax row stats over n for k channels: 512 rows = (b, ck)
// ---------------------------------------------------------------------------
__global__ void softmax_stats_kernel(const float* __restrict__ qkv,
                                     float* __restrict__ kmax, float* __restrict__ ksum) {
    int r = blockIdx.x;                                   // b*64 + ck
    int b = r >> 6, ck = r & 63;
    const float* row = qkv + (size_t)(CK0 + ck) * M_TOT + b * NN;
    __shared__ float red[256];
    float m = -__builtin_inff();
    for (int i = threadIdx.x; i < NN; i += 256) m = fmaxf(m, row[i]);
    red[threadIdx.x] = m; __syncthreads();
    for (int o = 128; o > 0; o >>= 1) {
        if (threadIdx.x < o) red[threadIdx.x] = fmaxf(red[threadIdx.x], red[threadIdx.x + o]);
        __syncthreads();
    }
    float mx = red[0]; __syncthreads();
    float s = 0.f;
    for (int i = threadIdx.x; i < NN; i += 256) s += __expf(row[i] - mx);
    red[threadIdx.x] = s; __syncthreads();
    for (int o = 128; o > 0; o >>= 1) {
        if (threadIdx.x < o) red[threadIdx.x] += red[threadIdx.x + o];
        __syncthreads();
    }
    if (threadIdx.x == 0) { kmax[r] = mx; ksum[r] = red[0]; }
}

// ---------------------------------------------------------------------------
// K6: content lambda: lamc[b][ki][vi] = sum_u sum_n softmax(k) * vbn
// ---------------------------------------------------------------------------
__global__ void lamc_kernel(const float* __restrict__ qkv,
                            const float* __restrict__ kmax, const float* __restrict__ ksum,
                            const float* __restrict__ scale, const float* __restrict__ shift,
                            float* __restrict__ lamc) {
    int blk = blockIdx.x;                                 // b*1024 + ki*64 + vi
    int b = blk >> 10, ki = (blk >> 6) & 15, vi = blk & 63;
    float tot = 0.f;
    for (int u = 0; u < UU; ++u) {
        int ck = ki * UU + u, cv = vi * UU + u;
        const float* krow = qkv + (size_t)(CK0 + ck) * M_TOT + b * NN;
        const float* vrow = qkv + (size_t)(CV0 + cv) * M_TOT + b * NN;
        float mx = kmax[b * 64 + ck];
        float inv = 1.f / ksum[b * 64 + ck];
        float sc = scale[192 + cv], sh = shift[192 + cv];
        float part = 0.f;
        for (int n = threadIdx.x; n < NN; n += 256)
            part += __expf(krow[n] - mx) * (vrow[n] * sc + sh);
        tot += part * inv;
    }
    __shared__ float red[256];
    red[threadIdx.x] = tot; __syncthreads();
    for (int o = 128; o > 0; o >>= 1) {
        if (threadIdx.x < o) red[threadIdx.x] += red[threadIdx.x + o];
        __syncthreads();
    }
    if (threadIdx.x == 0) lamc[blk] = red[0];
}

// ---------------------------------------------------------------------------
// K7: fused position-lambda conv + output einsums.
//     block = (b, 4 consecutive n); thread = (nn in 0..3, vi in 0..63)
//     Y[b,n,h*64+vi] = sum_k qbn[h,k,n] * (lamc[b,k,vi] + lamp[k,vi,n])
// ---------------------------------------------------------------------------
__global__ void final_kernel(const float* __restrict__ qkv,
                             const float* __restrict__ scale, const float* __restrict__ shift,
                             const float* __restrict__ lamc, const float* __restrict__ rel,
                             float* __restrict__ out) {
    __shared__ float s_rel[KK * UU * LL];                 // 1472 floats
    __shared__ float s_q[4][HH * KK];                     // 4 positions x 192 q-channels
    int b  = blockIdx.x >> 10;
    int n0 = (blockIdx.x & 1023) * 4;
    int tid = threadIdx.x;

    for (int i = tid; i < KK * UU * LL; i += 256) s_rel[i] = rel[i];
    for (int i = tid; i < 4 * HH * KK; i += 256) {
        int nn = i / (HH * KK), cq = i % (HH * KK);
        s_q[nn][cq] = qkv[(size_t)cq * M_TOT + b * NN + n0 + nn] * scale[cq] + shift[cq];
    }
    __syncthreads();

    int vi = tid & 63, nn = tid >> 6;
    int n = n0 + nn;

    float lam[KK];
#pragma unroll
    for (int k = 0; k < KK; ++k) lam[k] = lamc[b * 1024 + k * 64 + vi];

    // position lambda: lam[k] += sum_{u,l} rel[k,u,l] * vbn[vi*4+u, n+l-PAD]
    for (int u = 0; u < UU; ++u) {
        int cv = vi * UU + u;
        float sc = scale[192 + cv], sh = shift[192 + cv];
        const float* vrow = qkv + (size_t)(CV0 + cv) * M_TOT + b * NN;
#pragma unroll
        for (int l = 0; l < LL; ++l) {
            int np = n + l - PADW;
            float w = (np >= 0 && np < NN) ? (vrow[np] * sc + sh) : 0.f;
            const float* rp = s_rel + u * LL + l;         // rel[k][u][0][l] = k*92 + u*23 + l
#pragma unroll
            for (int k = 0; k < KK; ++k) lam[k] = fmaf(rp[k * UU * LL], w, lam[k]);
        }
    }

    float* orow = out + ((size_t)(b * NN + n)) * (HH * VV) + vi;
#pragma unroll
    for (int h = 0; h < HH; ++h) {
        float acc = 0.f;
        const float* qp = &s_q[nn][h * KK];
#pragma unroll
        for (int k = 0; k < KK; ++k) acc = fmaf(qp[k], lam[k], acc);
        orow[h * VV] = acc;
    }
}

// ---------------------------------------------------------------------------
// Host side
// ---------------------------------------------------------------------------
extern "C" void kernel_launch(void* const* d_in, const int* in_sizes, int n_in,
                              void* d_out, int out_size, void* d_ws, size_t ws_size,
                              hipStream_t stream) {
    const float* x    = (const float*)d_in[0];
    const float* Wq   = (const float*)d_in[1];
    const float* Wk   = (const float*)d_in[2];
    const float* Wv   = (const float*)d_in[3];
    const float* gq   = (const float*)d_in[4];
    const float* bq   = (const float*)d_in[5];
    const float* gv   = (const float*)d_in[6];
    const float* bv   = (const float*)d_in[7];
    const float* rel  = (const float*)d_in[8];
    float* out = (float*)d_out;

    // workspace layout (bytes)
    char* ws = (char*)d_ws;
    _Float16* xh  = (_Float16*)(ws + 0);                      // 32768*768*2   = 50,331,648
    _Float16* wh  = (_Float16*)(ws + 50331648);               // 512*768*2     =    786,432
    float* qkv    = (float*)(ws + 51118080);                  // 512*32768*4   = 67,108,864
    float* scale  = (float*)(ws + 118226944);                 // 448*4
    float* shift  = (float*)(ws + 118228736);                 // 448*4
    float* kmax   = (float*)(ws + 118230528);                 // 512*4
    float* ksum   = (float*)(ws + 118232576);                 // 512*4
    float* lamc   = (float*)(ws + 118234624);                 // 8*16*64*4 = 32768

    // 1-2: f32 -> f16 conversion / weight packing
    cvt_x_kernel<<<(M_TOT * DD) / 256, 256, 0, stream>>>(x, xh);
    cvt_w_kernel<<<(C_TOT * DD) / 256, 256, 0, stream>>>(Wq, Wk, Wv, wh);

    // 3: WMMA projection GEMM [512 x 768] x [768 x 32768]
    gemm_qkv_kernel<<<dim3(C_TOT / 64, M_TOT / 64), 128, 0, stream>>>(wh, xh, qkv);

    // 4: batchnorm folded scale/shift for q + v channels
    bn_stats_kernel<<<448, 256, 0, stream>>>(qkv, gq, bq, gv, bv, scale, shift);

    // 5: softmax row stats for k channels
    softmax_stats_kernel<<<512, 256, 0, stream>>>(qkv, kmax, ksum);

    // 6: content lambda
    lamc_kernel<<<BB * KK * VV, 256, 0, stream>>>(qkv, kmax, ksum, scale, shift, lamc);

    // 7: fused position lambda + output
    final_kernel<<<BB * (NN / 4), 256, 0, stream>>>(qkv, scale, shift, lamc, rel, out);
}
